// LMSPS_Se_721554506490
// MI455X (gfx1250) — matrix operand order repaired and checked by reading, not compiled, not run on hardware
//
#include <hip/hip_runtime.h>
#include <hip/hip_bf16.h>
#include <cstdint>
#include <math.h>

#define NROWS  100000
#define HID    256
#define NCLASS 64
#define TAU    1.0f
#define BN_EPS 1e-5f
#define HPAD   (HID + 8)   // padded bf16 row stride for h in LDS

// gfx1250 async global->LDS path (ASYNCcnt-tracked), with sync fallback
#if __has_builtin(__builtin_amdgcn_global_load_async_to_lds_b32) && \
    __has_builtin(__builtin_amdgcn_s_wait_asynccnt)
#define HAVE_ASYNC 1
typedef __attribute__((address_space(1))) int g_int;   // global int
typedef __attribute__((address_space(3))) int l_int;   // LDS int
#else
#define HAVE_ASYNC 0
#endif

typedef __attribute__((ext_vector_type(16))) __bf16 bf16x16;
typedef __attribute__((ext_vector_type(8)))  float  f32x8;

union Frag {
  uint32_t u[8];
  bf16x16  v;
};

__device__ __forceinline__ uint32_t f32_to_bf16_bits(float f) {
  uint32_t x = __builtin_bit_cast(uint32_t, f);
  x += 0x7FFFu + ((x >> 16) & 1u);   // round-to-nearest-even
  return x >> 16;
}
__device__ __forceinline__ uint32_t pack_bf16(float lo, float hi) {
  return (f32_to_bf16_bits(lo) & 0xFFFFu) | (f32_to_bf16_bits(hi) << 16);
}

struct Params {
  const float* feat[8];
  const float* emb[8];
  const float* w_out;
  const float* prelu;
  const float* probs;   // ws[0..7]
  float* gsum;          // ws[8..71]
  float* gsumsq;        // ws[72..135]
  float* z;             // d_out (unnormalized z, normalized in place later)
};

// ---------------------------------------------------------------------------
// Kernel 0: gumbel-softmax gate probs + zero BN accumulators
// ---------------------------------------------------------------------------
__global__ void gate_init(const float* __restrict__ alpha,
                          const float* __restrict__ gum,
                          const int*   __restrict__ samp,
                          float* __restrict__ probs,
                          float* __restrict__ gsum,
                          float* __restrict__ gsumsq) {
  int t = threadIdx.x;
  if (t < NCLASS) { gsum[t] = 0.0f; gsumsq[t] = 0.0f; }
  if (t == 0) {
    float wv[8]; float mx = -3.4e38f;
    for (int i = 0; i < 8; ++i) { wv[i] = alpha[samp[i]]; mx = fmaxf(mx, wv[i]); }
    float se = 0.0f;
    for (int i = 0; i < 8; ++i) se += expf(wv[i] - mx);
    float lse = mx + logf(se);
    float lg[8]; float mx2 = -3.4e38f;
    for (int i = 0; i < 8; ++i) { lg[i] = (wv[i] - lse + gum[i]) * (1.0f / TAU); mx2 = fmaxf(mx2, lg[i]); }
    float e[8]; float se2 = 0.0f;
    for (int i = 0; i < 8; ++i) { e[i] = expf(lg[i] - mx2); se2 += e[i]; }
    for (int i = 0; i < 8; ++i) probs[i] = e[i] / se2;
  }
}

// ---------------------------------------------------------------------------
// Raw A-tile staging: 64 rows x 32 k of fp32 (8 KB) into an LDS buffer.
// Async (GLOBAL_LOAD_ASYNC_TO_LDS_B32, 8 ops/thread -> ASYNCcnt=8/wave) or
// plain load/store fallback. b32 granularity because feature rows are only
// 4-byte aligned (D = 334, 745, ...).
// ---------------------------------------------------------------------------
__device__ __forceinline__ void stage_rawA(const float* __restrict__ F,
                                           long long rowBase, int D, int k0,
                                           float* buf, int tid) {
  #pragma unroll
  for (int i = 0; i < 8; ++i) {
    int g   = tid + 256 * i;      // 0..2047
    int row = g >> 5;             // 0..63
    int col = g & 31;             // 0..31
    long long r = rowBase + row;
    if (r >= NROWS) r = NROWS - 1;          // clamp: value unused for pad rows
    const float* gp = F + r * (long long)D + k0 + col;
#if HAVE_ASYNC
    __builtin_amdgcn_global_load_async_to_lds_b32(
        (g_int*)gp, (l_int*)(buf + row * 32 + col), 0, 0);
#else
    buf[row * 32 + col] = *gp;
#endif
  }
}

// ---------------------------------------------------------------------------
// Kernel 1: fused weighted multi-GEMM + PReLU + output head + BN partial sums
// Block: 256 threads = 8 wave32s, covers 64 rows x 256 hidden cols.
// Wave w: row-tile rg = w>>1 (16 rows), col-half cg = w&1 (8 x 16-col tiles).
// Double-buffered async A staging overlaps next chunk's HBM traffic with the
// current chunk's WMMAs.
// ---------------------------------------------------------------------------
__global__ __launch_bounds__(256, 2) void fused_main(Params p) {
  constexpr int KD[8] = {334, 512, 128, 745, 256, 600, 64, 64};

  __shared__ union SMem {
    struct {
      uint32_t sB[16 * 32 * 8];    // B fragments (32k x 256n)  16 KB
      float    rawA[2][64 * 32];   // raw fp32 A tiles          16 KB
      uint32_t sA[4 * 32 * 8];     // tail A fragments           4 KB
    } g1;
    uint16_t sH[64 * HPAD];        // hidden acts bf16         ~33 KB
  } smem;

  const int tid  = threadIdx.x;
  const int lane = tid & 31;
  const int wid  = tid >> 5;
  const int rg   = wid >> 1;             // 0..3
  const int cg   = wid & 1;              // 0..1
  const int lm   = lane & 15;
  const int lh   = lane >> 4;
  const long long rowBase = (long long)blockIdx.x * 64;

  f32x8 acc[8];
  #pragma unroll
  for (int t = 0; t < 8; ++t)
    #pragma unroll
    for (int j = 0; j < 8; ++j) acc[t][j] = 0.0f;

  // 8 WMMAs against the staged B fragments
  auto compute8 = [&](const Frag& a) {
    #pragma unroll
    for (int t = 0; t < 8; ++t) {
      Frag b;
      const uint32_t* bBase = &smem.g1.sB[(((cg << 3) + t) * 32 + lane) * 8];
      #pragma unroll
      for (int v = 0; v < 8; ++v) b.u[v] = bBase[v];
      acc[t] = __builtin_amdgcn_wmma_f32_16x16x32_bf16(
          false, a.v, false, b.v, (short)0, acc[t], false, false);
    }
  };

  // ---------------- gemm1: x = sum_c probs[c] * feat_c @ emb_c --------------
  #pragma unroll
  for (int c = 0; c < 8; ++c) {
    const float* __restrict__ F = p.feat[c];
    const float* __restrict__ E = p.emb[c];
    const float prob  = p.probs[c];
    const int   D     = KD[c];
    const int   NFULL = D >> 5;      // compile-time after unroll
    const int   TAIL  = D & 31;

    if (NFULL > 0) {
      __syncthreads();                           // buffers free
      stage_rawA(F, rowBase, D, 0, smem.g1.rawA[0], tid);
      #pragma unroll 1
      for (int ic = 0; ic < NFULL; ++ic) {
        const int k0 = ic << 5;
        __syncthreads();                         // compute(ic-1) done; sB free
        const bool more = (ic + 1 < NFULL);
        if (more)
          stage_rawA(F, rowBase, D, k0 + 32, smem.g1.rawA[(ic + 1) & 1], tid);
        // ---- stage B fragments (unmasked, pre-scaled by gate prob) ----
        #pragma unroll
        for (int i = 0; i < 16; ++i) {
          int g  = tid + 256 * i;                // (k-pair, n), coalesced in n
          int n  = g & 255;
          int kk = (g >> 8) << 1;
          int k  = k0 + kk;
          float b0 = E[(long long)k * HID + n]       * prob;
          float b1 = E[(long long)(k + 1) * HID + n] * prob;
          int fl = ((kk >> 4) << 4) | (n & 15);
          int fv = (kk >> 1) & 7;
          smem.g1.sB[((n >> 4) * 32 + fl) * 8 + fv] = pack_bf16(b0, b1);
        }
#if HAVE_ASYNC
        if (more) __builtin_amdgcn_s_wait_asynccnt(8);  // allow next chunk in flight
        else      __builtin_amdgcn_s_wait_asynccnt(0);
#endif
        __syncthreads();                         // async A + sB visible to all
        // ---- build A fragment straight from the raw fp32 tile ----
        const float* rb = smem.g1.rawA[ic & 1] + ((rg << 4) + lm) * 32;
        Frag a;
        #pragma unroll
        for (int v = 0; v < 8; ++v) {
          int kk = (lh << 4) | (v << 1);
          a.u[v] = pack_bf16(rb[kk], rb[kk + 1]);
        }
        compute8(a);
      }
    }

    if (TAIL) {   // one masked chunk; clamped loads + select, no exec branches
      const int k0 = NFULL << 5;
      __syncthreads();
      #pragma unroll
      for (int i = 0; i < 4; ++i) {
        int g  = tid + 256 * i;                  // (m, k-pair)
        int m  = g >> 4;
        int kk = (g & 15) << 1;
        long long row = rowBase + m;
        if (row >= NROWS) row = NROWS - 1;
        const float* src = F + row * (long long)D;
        int k  = k0 + kk;
        int kc0 = (k     < D) ? k     : (D - 1);
        int kc1 = (k + 1 < D) ? k + 1 : (D - 1);
        float a0 = src[kc0]; a0 = (k     < D) ? a0 : 0.0f;
        float a1 = src[kc1]; a1 = (k + 1 < D) ? a1 : 0.0f;
        int fl = ((kk >> 4) << 4) | (m & 15);
        int fv = (kk >> 1) & 7;
        smem.g1.sA[((m >> 4) * 32 + fl) * 8 + fv] = pack_bf16(a0, a1);
      }
      #pragma unroll
      for (int i = 0; i < 16; ++i) {
        int g  = tid + 256 * i;
        int n  = g & 255;
        int kk = (g >> 8) << 1;
        int k  = k0 + kk;
        int kc0 = (k     < D) ? k     : (D - 1);
        int kc1 = (k + 1 < D) ? k + 1 : (D - 1);
        float b0 = E[(long long)kc0 * HID + n] * prob; b0 = (k     < D) ? b0 : 0.0f;
        float b1 = E[(long long)kc1 * HID + n] * prob; b1 = (k + 1 < D) ? b1 : 0.0f;
        int fl = ((kk >> 4) << 4) | (n & 15);
        int fv = (kk >> 1) & 7;
        smem.g1.sB[((n >> 4) * 32 + fl) * 8 + fv] = pack_bf16(b0, b1);
      }
      __syncthreads();
      Frag a;
      const uint32_t* aBase = &smem.g1.sA[(rg * 32 + lane) * 8];
      #pragma unroll
      for (int v = 0; v < 8; ++v) a.u[v] = aBase[v];
      compute8(a);
    }
  }

  // ---------------- PReLU + restage h as bf16 in LDS (aliases sB/rawA) ------
  const float pa = *p.prelu;
  __syncthreads();
  #pragma unroll
  for (int t = 0; t < 8; ++t) {
    int hcol = (cg << 7) + (t << 4) + lm;        // 0..255
    #pragma unroll
    for (int v = 0; v < 8; ++v) {
      int m = (rg << 4) + (lh << 3) + v;         // 0..63
      float x = acc[t][v];
      x = (x >= 0.0f) ? x : pa * x;
      smem.sH[m * HPAD + hcol] = (uint16_t)f32_to_bf16_bits(x);
    }
  }
  __syncthreads();

  // ---------------- gemm2: z[64x64] = h @ w_out^T ---------------------------
  f32x8 zacc[2];
  #pragma unroll
  for (int j = 0; j < 2; ++j)
    #pragma unroll
    for (int v = 0; v < 8; ++v) zacc[j][v] = 0.0f;

  #pragma unroll 1
  for (int k0 = 0; k0 < HID; k0 += 32) {
    Frag a;
    int m = (rg << 4) + lm;
    #pragma unroll
    for (int v = 0; v < 8; ++v) {
      int kk = (lh << 4) | (v << 1);
      a.u[v] = *reinterpret_cast<const uint32_t*>(&smem.sH[m * HPAD + k0 + kk]);
    }
    #pragma unroll
    for (int j = 0; j < 2; ++j) {
      int cls = (((cg << 1) + j) << 4) + lm;     // 0..63
      Frag b;
      #pragma unroll
      for (int v = 0; v < 8; ++v) {
        int kk = (lh << 4) | (v << 1);
        const float* wp = p.w_out + cls * HID + k0 + kk;  // w_out^T, L2-hot
        b.u[v] = pack_bf16(wp[0], wp[1]);
      }
      zacc[j] = __builtin_amdgcn_wmma_f32_16x16x32_bf16(
          false, a.v, false, b.v, (short)0, zacc[j], false, false);
    }
  }

  // ---------------- epilogue: store z + BN partial sums ---------------------
  #pragma unroll
  for (int j = 0; j < 2; ++j) {
    int cls = (((cg << 1) + j) << 4) + lm;
    float s = 0.0f, s2 = 0.0f;
    #pragma unroll
    for (int v = 0; v < 8; ++v) {
      long long row = rowBase + (rg << 4) + (lh << 3) + v;
      float zv = zacc[j][v];
      if (row < NROWS) {
        p.z[row * NCLASS + cls] = zv;
        s  += zv;
        s2 += zv * zv;
      }
    }
    // lanes L and L+16 hold the same class, different rows -> fold, then atomic
    float so  = __shfl_down(s,  16, 32);
    float s2o = __shfl_down(s2, 16, 32);
    if (lane < 16) {
      atomicAdd(&p.gsum[cls],   s  + so);
      atomicAdd(&p.gsumsq[cls], s2 + s2o);
    }
  }
}

// ---------------------------------------------------------------------------
// Kernel 2: in-place BatchNorm normalization
// ---------------------------------------------------------------------------
__global__ __launch_bounds__(256) void bn_norm(const float* __restrict__ gsum,
                                               const float* __restrict__ gsumsq,
                                               float* __restrict__ out) {
  long long idx = (long long)blockIdx.x * 256 + threadIdx.x;
  if (idx >= (long long)NROWS * NCLASS) return;
  int c = (int)(idx & (NCLASS - 1));
  const float inv = 1.0f / (float)NROWS;
  float mean = gsum[c] * inv;
  float var  = gsumsq[c] * inv - mean * mean;
  out[idx] = (out[idx] - mean) * rsqrtf(var + BN_EPS);
}

// ---------------------------------------------------------------------------
extern "C" void kernel_launch(void* const* d_in, const int* in_sizes, int n_in,
                              void* d_out, int out_size, void* d_ws, size_t ws_size,
                              hipStream_t stream) {
  (void)in_sizes; (void)n_in; (void)out_size; (void)ws_size;
  Params p;
  // dict order: feat0,emb0,...,feat5,emb5, lab0,lemb0,lab1,lemb1,
  //             alpha,gumbels,prelu_a,w_out,epoch_sampled
  for (int i = 0; i < 6; ++i) {
    p.feat[i] = (const float*)d_in[2 * i];
    p.emb[i]  = (const float*)d_in[2 * i + 1];
  }
  p.feat[6] = (const float*)d_in[12]; p.emb[6] = (const float*)d_in[13];
  p.feat[7] = (const float*)d_in[14]; p.emb[7] = (const float*)d_in[15];
  const float* alpha = (const float*)d_in[16];
  const float* gum   = (const float*)d_in[17];
  p.prelu            = (const float*)d_in[18];
  p.w_out            = (const float*)d_in[19];
  const int* samp    = (const int*)d_in[20];

  float* ws = (float*)d_ws;
  p.probs  = ws;
  p.gsum   = ws + 8;
  p.gsumsq = ws + 8 + NCLASS;
  p.z      = (float*)d_out;

  gate_init<<<1, 64, 0, stream>>>(alpha, gum, samp, ws, p.gsum, p.gsumsq);

  int nblk = (NROWS + 63) / 64;                 // 1563
  fused_main<<<nblk, 256, 0, stream>>>(p);

  long long total = (long long)NROWS * NCLASS;  // 6.4M
  int nb2 = (int)((total + 255) / 256);
  bn_norm<<<nb2, 256, 0, stream>>>(p.gsum, p.gsumsq, (float*)d_out);
}
